// GlobalFluxTracker_13357348290703
// MI455X (gfx1250) — compile-verified, matrix operand order checked by reading, not run on hardware
//
#include <hip/hip_runtime.h>
#include <hip/hip_bf16.h>

// GlobalFluxTracker: u = xcat @ W_in^T + b_in ; h = complex diag scan(u) ;
// out = hcat @ W_out^T + b_out.  B=4, T=4096, D=1024 (2D=2048).
//
// GEMMs: f16 WMMA (v_wmma_f32_16x16x32_f16, f32 accum), W tile staged in LDS
// via async global->LDS loads (double buffered, ASYNCcnt).  Scan: f32.

#define TWO_D 2048
#define ONE_D 1024
#define BT    16384            // B*T
#define TLEN  4096
#define LDS_STRIDE 48          // f16 elements per LDS B-row (96B, 16B aligned, depads banks)

typedef __attribute__((ext_vector_type(16))) _Float16 v16h;
typedef __attribute__((ext_vector_type(8)))  _Float16 v8h;
typedef __attribute__((ext_vector_type(8)))  float    v8f;

// Fragment layout (ISA 7.12.2, 16-bit A/B, 16x(K=32) per wave):
// lane = (row|col)&15, hi = lane>>4; VGPR0..3 <- K = hi*8+0..7, VGPR4..7 <- K = 16+hi*8+0..7.
// base points at rowptr + kbase + hi*8 (16B aligned); second chunk at +16 elements.
__device__ __forceinline__ v16h load_frag(const _Float16* base) {
    v8h lo = *(const v8h*)(base);
    v8h hh = *(const v8h*)(base + 16);
    return __builtin_shufflevector(lo, hh, 0,1,2,3,4,5,6,7,8,9,10,11,12,13,14,15);
}

__device__ __forceinline__ v8f wmma_f16(v16h a, v16h b, v8f c) {
    return __builtin_amdgcn_wmma_f32_16x16x32_f16(false, a, false, b, (short)0, c,
                                                  false, false);
}

// ---------------------------------------------------------------------------
// GEMM: C[M,2048] = A[M,2048](f16) @ W[2048,2048]^T(f16) + bias, f32 out.
// Block = 256 threads = 8 waves; all waves share n_base (64 cols), each wave
// owns a 32-row M slice.  B tile (64 x 32 f16) staged in LDS via
// global_load_async_to_lds_b128, double buffered.
// Grid: (M/32)*(2048/64)/8 blocks (M=16384 -> 2048 blocks).
// ---------------------------------------------------------------------------
__global__ void __launch_bounds__(256)
gemm16_wmma(const _Float16* __restrict__ A,
            const _Float16* __restrict__ W,
            const float*    __restrict__ bias,
            float*          __restrict__ C,
            int M)
{
    __shared__ _Float16 ldsB[2][64 * LDS_STRIDE];   // 2 x 6144 B

    const int tid  = threadIdx.x;
    const int lane = tid & 31;
    const int wib  = tid >> 5;
    const int w    = blockIdx.x * 8 + wib;

    const int mTiles = M >> 5;
    const int mBlk   = w % mTiles;       // consecutive within a block
    const int nGrp   = w / mTiles;       // uniform within a block (512 % 8 == 0)
    const int m_base = mBlk * 32;
    const int n_base = nGrp * 64;

    const int r15 = lane & 15;
    const int hi  = lane >> 4;

    // ---- async-copy mapping: thread -> (row n = tid>>2, 16B chunk c = tid&3)
    const int cp_n = tid >> 2;
    const int cp_c = tid & 3;
    const unsigned gW_byte0 = ((unsigned)(n_base + cp_n) * TWO_D + (unsigned)cp_c * 8) * 2u;
    const unsigned lds_dst0 = (unsigned)(size_t)&ldsB[0][(unsigned)cp_n * LDS_STRIDE + cp_c * 8];
    const unsigned lds_dst1 = (unsigned)(size_t)&ldsB[1][(unsigned)cp_n * LDS_STRIDE + cp_c * 8];
    const unsigned long long Wbase = (unsigned long long)(size_t)W;

    // ---- A row pointers (per-lane, global)
    const _Float16* Arow0 = A + (size_t)(m_base + r15) * TWO_D + hi * 8;
    const _Float16* Arow1 = Arow0 + (size_t)16 * TWO_D;

    // ---- LDS read base for B fragments (row n = nt*16 + r15)
    const _Float16* lb0 = &ldsB[0][(unsigned)r15 * LDS_STRIDE + hi * 8];
    const _Float16* lb1 = &ldsB[1][(unsigned)r15 * LDS_STRIDE + hi * 8];

    v8f acc[2][4];
#pragma unroll
    for (int mt = 0; mt < 2; ++mt)
#pragma unroll
        for (int nt = 0; nt < 4; ++nt) {
            float bv = bias[n_base + nt * 16 + r15];
#pragma unroll
            for (int e = 0; e < 8; ++e) acc[mt][nt][e] = bv;
        }

    // issue one 16B async global->LDS copy per thread (4KB tile per call)
    auto issue = [&](unsigned ldsd, int kb) {
        unsigned goff = gW_byte0 + (unsigned)kb * 2u;
        asm volatile("global_load_async_to_lds_b128 %0, %1, %2"
                     :: "v"(ldsd), "v"(goff), "s"(Wbase) : "memory");
    };

    issue(lds_dst0, 0);
    issue(lds_dst1, 32);

    int cur = 0;
    for (int kb = 0; kb < TWO_D; kb += 32) {
        // oldest of the two in-flight tiles (the one we read now) is complete
        asm volatile("s_wait_asynccnt 0x1" ::: "memory");
        __syncthreads();

        __builtin_prefetch(Arow0 + kb + 64, 0, 3);
        __builtin_prefetch(Arow1 + kb + 64, 0, 3);

        v16h a0 = load_frag(Arow0 + kb);
        v16h a1 = load_frag(Arow1 + kb);
        const _Float16* lbc = cur ? lb1 : lb0;
#pragma unroll
        for (int nt = 0; nt < 4; ++nt) {
            v16h bf = load_frag(lbc + nt * 16 * LDS_STRIDE);
            acc[0][nt] = wmma_f16(a0, bf, acc[0][nt]);
            acc[1][nt] = wmma_f16(a1, bf, acc[1][nt]);
        }

        __syncthreads();   // all waves done reading buffer `cur`
        // refill `cur` for kb+64; wraps mod K in the last two iterations
        // (dummy data, never read; keeps in-flight count uniform at 2)
        issue(cur ? lds_dst1 : lds_dst0, (kb + 64) & (TWO_D - 1));
        cur ^= 1;
    }

    // C/D layout: lanes 0-15 -> M rows 0..7 (vgpr idx), lanes 16-31 -> rows 8..15
#pragma unroll
    for (int mt = 0; mt < 2; ++mt) {
        const int mrow = m_base + mt * 16 + 8 * hi;
#pragma unroll
        for (int nt = 0; nt < 4; ++nt) {
            const int n = n_base + nt * 16 + r15;
            float* cp = C + (size_t)mrow * TWO_D + n;
#pragma unroll
            for (int r = 0; r < 8; ++r)
                cp[(size_t)r * TWO_D] = acc[mt][nt][r];
        }
    }
}

// ---------------------------------------------------------------------------
__global__ void cvt_f16(const float* __restrict__ src, _Float16* __restrict__ dst, int n)
{
    int i = blockIdx.x * blockDim.x + threadIdx.x;
    if (i < n) dst[i] = (_Float16)src[i];
}

// xcat_f16[m, 0:1024] = x_re[m,:] ; xcat_f16[m, 1024:2048] = x_im[m,:]
__global__ void pack_x(const float* __restrict__ xr,
                       const float* __restrict__ xi,
                       _Float16* __restrict__ xcat)
{
    size_t i = (size_t)blockIdx.x * blockDim.x + threadIdx.x;   // < BT*ONE_D
    size_t m = i >> 10;
    size_t d = i & 1023;
    xcat[m * TWO_D + d]         = (_Float16)xr[i];
    xcat[m * TWO_D + ONE_D + d] = (_Float16)xi[i];
}

// complex diagonal scan: h_t = decay*h_{t-1} + u_t ; decay = sigmoid(re)+i*im
__global__ void scan_kernel(const float* __restrict__ u,      // [BT, 2048] f32
                            const float* __restrict__ dre,    // [1024]
                            const float* __restrict__ dim_,   // [1024]
                            _Float16*    __restrict__ hcat)   // [BT, 2048] f16
{
    int c = blockIdx.x * blockDim.x + threadIdx.x;   // 0..4095
    int b = c >> 10;
    int d = c & 1023;
    float ar = 1.0f / (1.0f + expf(-dre[d]));
    float ai = dim_[d];
    float hr = 0.0f, hm = 0.0f;

    size_t rowoff = (size_t)b * TLEN * TWO_D + d;
    for (int t = 0; t < TLEN; ++t) {
        const float* up = u + rowoff + (size_t)t * TWO_D;
        float ur = up[0];
        float ui = up[ONE_D];
        float nr = ar * hr - ai * hm + ur;
        float ni = ar * hm + ai * hr + ui;
        hr = nr; hm = ni;
        _Float16* hp = hcat + rowoff + (size_t)t * TWO_D;
        hp[0]     = (_Float16)hr;
        hp[ONE_D] = (_Float16)hm;
    }
}

// ---------------------------------------------------------------------------
extern "C" void kernel_launch(void* const* d_in, const int* in_sizes, int n_in,
                              void* d_out, int out_size, void* d_ws, size_t ws_size,
                              hipStream_t stream)
{
    const float* x_re     = (const float*)d_in[0];   // [4,4096,1024]
    const float* x_im     = (const float*)d_in[1];
    const float* decay_re = (const float*)d_in[2];   // [1024]
    const float* decay_im = (const float*)d_in[3];
    const float* W_in     = (const float*)d_in[4];   // [2048,2048]
    const float* b_in     = (const float*)d_in[5];   // [2048]
    const float* W_out    = (const float*)d_in[6];
    const float* b_out    = (const float*)d_in[7];
    float* out = (float*)d_out;                      // [4,4096,2048]

    char* ws = (char*)d_ws;
    float*    u_cat  = (float*)   (ws);                           // 134,217,728 B
    _Float16* xcat   = (_Float16*)(ws + 134217728ull);            //  67,108,864 B
    _Float16* hcat   = (_Float16*)(ws + 201326592ull);            //  67,108,864 B
    _Float16* Win16  = (_Float16*)(ws + 268435456ull);            //   8,388,608 B
    _Float16* Wout16 = (_Float16*)(ws + 276824064ull);            //   8,388,608 B

    cvt_f16<<<16384, 256, 0, stream>>>(W_in,  Win16,  TWO_D * TWO_D);
    cvt_f16<<<16384, 256, 0, stream>>>(W_out, Wout16, TWO_D * TWO_D);
    pack_x <<<65536, 256, 0, stream>>>(x_re, x_im, xcat);

    // u_cat = xcat @ W_in^T + b_in
    gemm16_wmma<<<2048, 256, 0, stream>>>(xcat, Win16, b_in, u_cat, BT);

    // h = scan(u)
    scan_kernel<<<16, 256, 0, stream>>>(u_cat, decay_re, decay_im, hcat);

    // out = hcat @ W_out^T + b_out
    gemm16_wmma<<<2048, 256, 0, stream>>>(hcat, Wout16, b_out, out, BT);
}
